// LearnableRAGGNN_69767448756415
// MI455X (gfx1250) — matrix operand order repaired and checked by reading, not compiled, not run on hardware
//
#include <hip/hip_runtime.h>
#include <hip/hip_bf16.h>
#include <math.h>

// ---------------------------------------------------------------------------
// Types for CDNA5 WMMA (wave32): A/B fragments are v16h (16 f16 per lane),
// C/D accumulators are v8f (8 f32 per lane) for 16x16 tiles.
// ---------------------------------------------------------------------------
typedef __attribute__((ext_vector_type(16))) _Float16 v16h;
typedef __attribute__((ext_vector_type(8)))  _Float16 h8;
typedef __attribute__((ext_vector_type(4)))  _Float16 h4;
typedef __attribute__((ext_vector_type(8)))  float    v8f;

#define GN    8192
#define GHID  1024
#define GD    256
#define GNDOC 16384
#define GK    32

// GEMM tiling: 128x128 block tile, K-step 32, 256 threads = 8 wave32.
#define BM 128
#define BN 128
#define BK 32
#define PAD 8   // f16 elements of row padding; row stride 40*2=80B (16B-aligned)

// ---------------------------------------------------------------------------
// WMMA f16 GEMM:  C[M,N] = A[M,K] * B  (+ bias[N]) (+= if ACCUM)
//   TRANS_B == false : B is [K,N] row-major with leading dim ldb
//   TRANS_B == true  : B is a weight W[N, ldb] row-major; effective
//                      B[k][n] = W[n][kofs + k]  (i.e. C = A @ W^T, K-slice)
// M % 128 == 0, N % 128 == 0, K % 32 == 0 (true for all calls below).
//
// Software pipeline per K-step:
//   (a) issue global loads for tile k+1 into registers (no wait)
//   (b) 8 WMMAs for tile k out of LDS (B-fragment double-rotated so each
//       WMMA does not drain DScnt)
//   (c) wait loads, fp32->f16 convert, ds_store into the other LDS buffer
// ---------------------------------------------------------------------------
template <bool TRANS_B, bool ACCUM>
__global__ __launch_bounds__(256)
void gemm_wmma(const float* __restrict__ A, const float* __restrict__ B,
               float* __restrict__ C, const float* __restrict__ bias,
               int M, int N, int K, int lda, int ldb, int kofs, int ldc)
{
    // A tile: [row M][col K]   (M-major, K contiguous)
    // B tile: [col N][row K]   (N-major, K contiguous -> fragment-major)
    __shared__ _Float16 As[2][BM][BK + PAD];
    __shared__ _Float16 Bs[2][BN][BK + PAD];

    constexpr int AIT  = (BM * BK) / (256 * 4);  // 4  float4 per thread
    constexpr int BIT4 = (BK * BN) / (256 * 4);  // 4  float4 per thread (trans)
    constexpr int BIT1 = (BK * BN) / 256;        // 16 floats per thread (no-trans)

    const int tid  = threadIdx.x;
    const int wave = tid >> 5;
    const int lane = tid & 31;
    const int half = lane >> 4;   // 0: lanes 0-15, 1: lanes 16-31
    const int r    = lane & 15;

    const int m0 = blockIdx.y * BM;
    const int n0 = blockIdx.x * BN;

    v8f acc[BN / 16];
#pragma unroll
    for (int t = 0; t < BN / 16; ++t)
#pragma unroll
        for (int j = 0; j < 8; ++j) acc[t][j] = 0.0f;

    float4 areg[AIT];
    float4 breg4[TRANS_B ? BIT4 : 1];
    float  breg1[TRANS_B ? 1 : BIT1];

    // ---- (a) issue global loads for tile at k0 into registers --------------
    auto load_regs = [&](int k0) {
#pragma unroll
        for (int i = 0; i < AIT; ++i) {
            const int idx = (i * 256 + tid) * 4;
            const int row = idx >> 5;        // /BK
            const int col = idx & (BK - 1);  // multiple of 4
            areg[i] = *reinterpret_cast<const float4*>(
                A + (long)(m0 + row) * lda + k0 + col);
        }
        if constexpr (TRANS_B) {
#pragma unroll
            for (int i = 0; i < BIT4; ++i) {
                const int idx = (i * 256 + tid) * 4;
                const int n   = idx >> 5;        // 0..127
                const int kc  = idx & (BK - 1);  // multiple of 4
                breg4[i] = *reinterpret_cast<const float4*>(
                    B + (long)(n0 + n) * ldb + kofs + k0 + kc);
            }
        } else {
#pragma unroll
            for (int i = 0; i < BIT1; ++i) {
                const int e = i * 256 + tid;
                const int k = e >> 7;        // 0..31
                const int n = e & (BN - 1);  // lane-consecutive -> coalesced
                breg1[i] = B[(long)(k0 + k) * ldb + n0 + n];
            }
        }
    };

    // ---- (c) convert staged registers and store to LDS buffer --------------
    auto store_tile = [&](int buf) {
#pragma unroll
        for (int i = 0; i < AIT; ++i) {
            const int idx = (i * 256 + tid) * 4;
            const int row = idx >> 5;
            const int col = idx & (BK - 1);
            h4 hv; hv[0] = (_Float16)areg[i].x; hv[1] = (_Float16)areg[i].y;
                   hv[2] = (_Float16)areg[i].z; hv[3] = (_Float16)areg[i].w;
            *reinterpret_cast<h4*>(&As[buf][row][col]) = hv;
        }
        if constexpr (TRANS_B) {
#pragma unroll
            for (int i = 0; i < BIT4; ++i) {
                const int idx = (i * 256 + tid) * 4;
                const int n   = idx >> 5;
                const int kc  = idx & (BK - 1);
                h4 hv; hv[0] = (_Float16)breg4[i].x; hv[1] = (_Float16)breg4[i].y;
                       hv[2] = (_Float16)breg4[i].z; hv[3] = (_Float16)breg4[i].w;
                *reinterpret_cast<h4*>(&Bs[buf][n][kc]) = hv;
            }
        } else {
#pragma unroll
            for (int i = 0; i < BIT1; ++i) {
                const int e = i * 256 + tid;
                const int k = e >> 7;
                const int n = e & (BN - 1);
                Bs[buf][n][k] = (_Float16)breg1[i];
            }
        }
    };

    // prologue: stage tile 0
    load_regs(0);
    store_tile(0);
    __syncthreads();

    const int arow = wave * 16 + r;
    const int nk = K / BK;
    for (int kt = 0; kt < nk; ++kt) {
        const int buf = kt & 1;
        const bool more = (kt + 1 < nk);
        if (more) load_regs((kt + 1) * BK);   // (a) loads in flight

        // ---- (b) compute from LDS buffer `buf` -----------------------------
        // A fragment (16x32): lanes 0-15 row M=r hold K {0..7,16..23},
        // lanes 16-31 hold K {8..15,24..31}  (ISA 7.12.2)
        const h8 alo = *reinterpret_cast<const h8*>(&As[buf][arow][half * 8]);
        const h8 ahi = *reinterpret_cast<const h8*>(&As[buf][arow][16 + half * 8]);
        v16h afrag;
#pragma unroll
        for (int e = 0; e < 8; ++e) { afrag[e] = alo[e]; afrag[8 + e] = ahi[e]; }

        // B fragment rotation: issue loads for tile t+1 before WMMA t
        h8 blo = *reinterpret_cast<const h8*>(&Bs[buf][r][half * 8]);
        h8 bhi = *reinterpret_cast<const h8*>(&Bs[buf][r][16 + half * 8]);
#pragma unroll
        for (int t = 0; t < BN / 16; ++t) {
            h8 nlo, nhi;
            if (t < BN / 16 - 1) {
                nlo = *reinterpret_cast<const h8*>(&Bs[buf][(t + 1) * 16 + r][half * 8]);
                nhi = *reinterpret_cast<const h8*>(&Bs[buf][(t + 1) * 16 + r][16 + half * 8]);
            }
            v16h bfrag;
#pragma unroll
            for (int e = 0; e < 8; ++e) { bfrag[e] = blo[e]; bfrag[8 + e] = bhi[e]; }
            acc[t] = __builtin_amdgcn_wmma_f32_16x16x32_f16(
                false, afrag, false, bfrag, (short)0, acc[t], false, false);
            if (t < BN / 16 - 1) { blo = nlo; bhi = nhi; }
        }

        // ---- (c) drain staged loads into the other buffer ------------------
        if (more) store_tile(1 - buf);
        __syncthreads();
    }

    // ---- epilogue: C 16x16 f32 layout -> VGPR v holds M = half*8 + v --------
#pragma unroll
    for (int t = 0; t < BN / 16; ++t) {
        const int col = n0 + t * 16 + r;
        const float bv = bias ? bias[col] : 0.0f;
#pragma unroll
        for (int v = 0; v < 8; ++v) {
            const int row = m0 + wave * 16 + half * 8 + v;
            float val = acc[t][v] + bv;
            float* p = C + (long)row * ldc + col;
            if (ACCUM) *p += val; else *p = val;
        }
    }
}

// ---------------------------------------------------------------------------
// Row LayerNorm + exact GELU, in place. One block (256 thr) per row of HID.
// ---------------------------------------------------------------------------
__device__ __forceinline__ float block_sum(float v, float* red)
{
    const int tid = threadIdx.x;
    red[tid] = v; __syncthreads();
    for (int off = 128; off > 0; off >>= 1) {
        if (tid < off) red[tid] += red[tid + off];
        __syncthreads();
    }
    const float r = red[0]; __syncthreads();
    return r;
}

__global__ __launch_bounds__(256)
void ln_gelu_kernel(float* __restrict__ X, const float* __restrict__ g,
                    const float* __restrict__ b, int H)
{
    __shared__ float red[256];
    float* x = X + (long)blockIdx.x * H;

    float s = 0.0f;
    for (int i = threadIdx.x; i < H; i += 256) s += x[i];
    const float mean = block_sum(s, red) / (float)H;

    float v = 0.0f;
    for (int i = threadIdx.x; i < H; i += 256) {
        const float d = x[i] - mean; v += d * d;
    }
    const float var = block_sum(v, red) / (float)H;
    const float inv = rsqrtf(var + 1e-5f);

    for (int i = threadIdx.x; i < H; i += 256) {
        const float y = (x[i] - mean) * inv * g[i] + b[i];
        x[i] = 0.5f * y * (1.0f + erff(y * 0.70710678118654752f));
    }
}

__global__ __launch_bounds__(256)
void gelu_kernel(float* __restrict__ X, long n)
{
    const long i = (long)blockIdx.x * 256 + threadIdx.x;
    if (i < n) {
        const float y = X[i];
        X[i] = 0.5f * y * (1.0f + erff(y * 0.70710678118654752f));
    }
}

// ---------------------------------------------------------------------------
// Per-row: scores = (q . doc)/sqrt(D); top-32; softmax(v/0.5); ctx gather.
// All 16384 scores kept in LDS (64KB dyn shared; CDNA5 WGP LDS = 320KB).
// ---------------------------------------------------------------------------
__global__ __launch_bounds__(256)
void topk_ctx_kernel(const float* __restrict__ q, const float* __restrict__ doc,
                     float* __restrict__ ctx)
{
    extern __shared__ float smem[];
    float* sc   = smem;                         // [GNDOC]
    float* sq   = sc + GNDOC;                   // [GD]
    float* redv = sq + GD;                      // [256]
    int*   redi = (int*)(redv + 256);           // [256]
    float* topv = (float*)(redi + 256);         // [GK]
    int*   topi = (int*)(topv + GK);            // [GK]

    const int row = blockIdx.x;
    const int tid = threadIdx.x;

    for (int i = tid; i < GD; i += 256) sq[i] = q[(long)row * GD + i];
    __syncthreads();

    // scores (1/sqrt(256) = 1/16), float4-vectorized dot products
    const float4* sq4 = reinterpret_cast<const float4*>(sq);
    for (int d = tid; d < GNDOC; d += 256) {
        const float4* dp = reinterpret_cast<const float4*>(doc + (long)d * GD);
        float s = 0.0f;
#pragma unroll 4
        for (int i = 0; i < GD / 4; ++i) {
            const float4 a = sq4[i];
            const float4 bv = dp[i];
            s += a.x * bv.x + a.y * bv.y + a.z * bv.z + a.w * bv.w;
        }
        sc[d] = s * 0.0625f;
    }
    __syncthreads();

    // 32 rounds of masked block-argmax
    for (int kk = 0; kk < GK; ++kk) {
        float bv = -3.4e38f; int bi = 0;
        for (int d = tid; d < GNDOC; d += 256)
            if (sc[d] > bv) { bv = sc[d]; bi = d; }
        redv[tid] = bv; redi[tid] = bi;
        __syncthreads();
        for (int off = 128; off > 0; off >>= 1) {
            if (tid < off && redv[tid + off] > redv[tid]) {
                redv[tid] = redv[tid + off]; redi[tid] = redi[tid + off];
            }
            __syncthreads();
        }
        if (tid == 0) {
            topv[kk] = redv[0]; topi[kk] = redi[0];
            sc[redi[0]] = -3.4e38f;
        }
        __syncthreads();
    }

    // softmax over top-K with temperature 0.5 (i.e. *2); topv[0] is the max
    const float mx = topv[0];
    if (tid < GK) topv[tid] = __expf((topv[tid] - mx) * 2.0f);
    __syncthreads();
    if (tid == 0) {
        float s = 0.0f;
        for (int k = 0; k < GK; ++k) s += topv[k];
        redv[0] = 1.0f / s;
    }
    __syncthreads();
    const float inv = redv[0];

    for (int i = tid; i < GD; i += 256) {
        float s = 0.0f;
#pragma unroll
        for (int k = 0; k < GK; ++k)
            s += topv[k] * doc[(long)topi[k] * GD + i];
        ctx[(long)row * GD + i] = s * inv;
    }
}

// ---------------------------------------------------------------------------
// fused = sigmoid(glin) * H + (1 - sigmoid(glin)) * clin
// ---------------------------------------------------------------------------
__global__ __launch_bounds__(256)
void fuse_kernel(const float* __restrict__ H, const float* __restrict__ glin,
                 const float* __restrict__ clin, float* __restrict__ out, long n)
{
    const long i = (long)blockIdx.x * 256 + threadIdx.x;
    if (i < n) {
        const float g = 1.0f / (1.0f + __expf(-glin[i]));
        out[i] = g * H[i] + (1.0f - g) * clin[i];
    }
}

// ---------------------------------------------------------------------------
// Host orchestration
// ---------------------------------------------------------------------------
static inline void launch_gemm(bool trans_b, bool accum,
                               const float* A, const float* B, float* C,
                               const float* bias, int M, int N, int K,
                               int lda, int ldb, int kofs, int ldc,
                               hipStream_t s)
{
    dim3 grid(N / BN, M / BM), blk(256);
    if (trans_b) {
        if (accum)
            gemm_wmma<true, true><<<grid, blk, 0, s>>>(A, B, C, bias, M, N, K, lda, ldb, kofs, ldc);
        else
            gemm_wmma<true, false><<<grid, blk, 0, s>>>(A, B, C, bias, M, N, K, lda, ldb, kofs, ldc);
    } else {
        gemm_wmma<false, false><<<grid, blk, 0, s>>>(A, B, C, bias, M, N, K, lda, ldb, kofs, ldc);
    }
}

extern "C" void kernel_launch(void* const* d_in, const int* in_sizes, int n_in,
                              void* d_out, int out_size, void* d_ws, size_t ws_size,
                              hipStream_t stream)
{
    (void)in_sizes; (void)n_in; (void)out_size; (void)ws_size;

    const float* A    = (const float*)d_in[0];
    const float* X    = (const float*)d_in[1];
    const float* doc  = (const float*)d_in[2];
    const float* W1   = (const float*)d_in[3];
    const float* W2   = (const float*)d_in[4];
    const float* W3   = (const float*)d_in[5];
    const float* g1   = (const float*)d_in[6];
    const float* b1   = (const float*)d_in[7];
    const float* g2   = (const float*)d_in[8];
    const float* b2   = (const float*)d_in[9];
    const float* g3   = (const float*)d_in[10];
    const float* b3   = (const float*)d_in[11];
    const float* pW1  = (const float*)d_in[12];
    const float* pb1  = (const float*)d_in[13];
    const float* pW2  = (const float*)d_in[14];
    const float* pb2  = (const float*)d_in[15];
    const float* gW   = (const float*)d_in[16];
    const float* gb   = (const float*)d_in[17];
    const float* cW   = (const float*)d_in[18];
    const float* cb   = (const float*)d_in[19];
    float* out = (float*)d_out;

    // workspace layout (bytes)
    char* ws = (char*)d_ws;
    const size_t SZ_NH = (size_t)GN * GHID * sizeof(float);   // 32 MB
    const size_t SZ_ND = (size_t)GN * GD   * sizeof(float);   //  8 MB
    float* ws0  = (float*)(ws);                               // T / qmid / clin
    float* ws1  = (float*)(ws + SZ_NH);                       // H
    float* wsQ  = (float*)(ws + 2 * SZ_NH);                   // q
    float* wsC  = (float*)(ws + 2 * SZ_NH + SZ_ND);           // ctx
    float* wsG  = (float*)(ws + 2 * SZ_NH + 2 * SZ_ND);       // glin

    const long NH = (long)GN * GHID;
    const dim3 eb(256);
    const int egrid = (int)((NH + 255) / 256);

    // --- 3-layer GCN: H = gelu(LN((A@H) @ W^T)) -----------------------------
    launch_gemm(false, false, A,   X,   ws0, nullptr, GN, GHID, GN,   GN,   GHID, 0, GHID, stream);
    launch_gemm(true,  false, ws0, W1,  ws1, nullptr, GN, GHID, GHID, GHID, GHID, 0, GHID, stream);
    ln_gelu_kernel<<<GN, eb, 0, stream>>>(ws1, g1, b1, GHID);

    launch_gemm(false, false, A,   ws1, ws0, nullptr, GN, GHID, GN,   GN,   GHID, 0, GHID, stream);
    launch_gemm(true,  false, ws0, W2,  ws1, nullptr, GN, GHID, GHID, GHID, GHID, 0, GHID, stream);
    ln_gelu_kernel<<<GN, eb, 0, stream>>>(ws1, g2, b2, GHID);

    launch_gemm(false, false, A,   ws1, ws0, nullptr, GN, GHID, GN,   GN,   GHID, 0, GHID, stream);
    launch_gemm(true,  false, ws0, W3,  ws1, nullptr, GN, GHID, GHID, GHID, GHID, 0, GHID, stream);
    ln_gelu_kernel<<<GN, eb, 0, stream>>>(ws1, g3, b3, GHID);
    // H lives in ws1

    // --- projection to query space: q = gelu(H@pW1^T+pb1) @ pW2^T + pb2 -----
    launch_gemm(true, false, ws1, pW1, ws0, pb1, GN, GHID, GHID, GHID, GHID, 0, GHID, stream);
    gelu_kernel<<<egrid, eb, 0, stream>>>(ws0, NH);
    launch_gemm(true, false, ws0, pW2, wsQ, pb2, GN, GD, GHID, GHID, GHID, 0, GD, stream);

    // --- top-K retrieval + softmax + context --------------------------------
    const size_t smem = (size_t)(GNDOC + GD + 256 + 256 + GK + GK) * 4;
    topk_ctx_kernel<<<GN, eb, smem, stream>>>(wsQ, doc, wsC);

    // --- gate: glin = [H, ctx] @ gW^T + gb  (two K-slices of gW) ------------
    launch_gemm(true, false, ws1, gW, wsG, gb,      GN, GHID, GHID, GHID, GHID + GD, 0,    GHID, stream);
    launch_gemm(true, true,  wsC, gW, wsG, nullptr, GN, GHID, GD,   GD,   GHID + GD, GHID, GHID, stream);

    // --- clin = ctx @ cW^T + cb  (reuse ws0) --------------------------------
    launch_gemm(true, false, wsC, cW, ws0, cb, GN, GHID, GD, GD, GD, 0, GHID, stream);

    // --- fused output -------------------------------------------------------
    fuse_kernel<<<egrid, eb, 0, stream>>>(ws1, wsG, ws0, out, NH);
}
